// Decoder_12807592477272
// MI455X (gfx1250) — compile-verified, matrix operand order checked by reading, not run on hardware
//
#include <hip/hip_runtime.h>
#include <math.h>

typedef __attribute__((ext_vector_type(16))) __bf16       v16bf;
typedef __attribute__((ext_vector_type(8)))  float        v8f;
typedef __attribute__((ext_vector_type(4)))  unsigned int u32x4;

constexpr int Bv = 32, Tv = 2048, Dv = 256, Hv = 1024, OUT2v = 512, TWOD = 512;
constexpr int ROWS = 64;   // query rows per block
constexpr int NW   = 4;    // waves per block (wave32)
constexpr int NT   = 128;  // threads per block

// ---- workspace layout (bf16 elements) ----
constexpr long WS_XB  = 0;                         // x  [B][T][D]        bf16
constexpr long WS_XT  = WS_XB + (long)Bv*Tv*Dv;    // xT [B][D][T]        bf16
constexpr long WS_W1T = WS_XT + (long)Bv*Tv*Dv;    // W1t [H][2D]         bf16
constexpr long WS_W2T = WS_W1T + (long)TWOD*Hv;    // W2t [OUT2][H]       bf16
constexpr long WS_END = WS_W2T + (long)Hv*OUT2v;   // ~34.6M elems = 69MB

// ---- LDS layout (bytes) ----
constexpr int OFF_DEC  = 0;                          // 64 x 512  bf16 (x | ctx)
constexpr int OFF_H    = OFF_DEC  + ROWS * TWOD * 2; // 64 x 1024 bf16 (hidden)
constexpr int OFF_XKR0 = OFF_H    + ROWS * Hv * 2;   // 32 x 256  bf16 keys row-major, buf0
constexpr int OFF_XKC0 = OFF_XKR0 + 32 * Dv * 2;     // 256 x 32  bf16 keys col-major, buf0
constexpr int OFF_XKR1 = OFF_XKC0 + Dv * 32 * 2;     // buf1
constexpr int OFF_XKC1 = OFF_XKR1 + 32 * Dv * 2;     // buf1
constexpr int OFF_P    = OFF_XKC1 + Dv * 32 * 2;     // 4 x 16x32 bf16 (P per wave)
constexpr int OFF_WA   = OFF_P    + NW * 16 * 32 * 2;// 256 x 32  bf16 weight tile buf A
constexpr int OFF_WB   = OFF_WA   + 256 * 32 * 2;    // weight tile buf B
constexpr int SMEM_BYTES = OFF_WB + 256 * 32 * 2;    // 299008 < 320KB

static __device__ inline v8f v8f_zero() {
  v8f z = {0.f, 0.f, 0.f, 0.f, 0.f, 0.f, 0.f, 0.f};
  return z;
}

// float -> bf16 bits, round-to-nearest-even
static __device__ inline unsigned short f2bf(float f) {
  unsigned int u = __float_as_uint(f);
  unsigned int r = (u + 0x7FFFu + ((u >> 16) & 1u)) >> 16;
  return (unsigned short)r;
}

static __device__ inline v8f wmma_bf16(v16bf a, v16bf b, v8f c) {
  // D = A(16x32 bf16) * B(32x16 bf16) + C(16x16 f32)
  return __builtin_amdgcn_wmma_f32_16x16x32_bf16(false, a, false, b, (short)0, c,
                                                 false, false);
}

// 16-byte async global->LDS copy per lane (ASYNCcnt tracked, no VGPR data path)
static __device__ inline void async_b128(unsigned lds_byte, const void* g) {
  asm volatile("global_load_async_to_lds_b128 %0, %1, off"
               :: "v"(lds_byte), "v"((unsigned long long)(size_t)g)
               : "memory");
}
static __device__ inline void wait_async0() {
  asm volatile("s_wait_asynccnt 0x0" ::: "memory");
}

// Load a 16x32 bf16 fragment (A layout; B layout mirrors with rows indexed by N)
// from a row-major LDS tile.
static __device__ inline v16bf load_frag(const unsigned short* base, int rowStride,
                                         int kbase, int lane) {
  int rl = lane & 15, hf = lane >> 4;
  const unsigned short* p = base + rl * rowStride + kbase + hf * 8;
  union { v16bf v; u32x4 q[2]; } u;
  u.q[0] = *(const u32x4*)(p);       // ds_load_b128
  u.q[1] = *(const u32x4*)(p + 16);  // ds_load_b128
  return u.v;
}

// ======================= prep kernels: f32 -> bf16 (+transpose) =======================
__global__ void conv_x_kernel(const float* __restrict__ x,
                              unsigned short* __restrict__ xb,
                              unsigned short* __restrict__ xT) {
  long stride = (long)gridDim.x * blockDim.x;
  for (long i = (long)blockIdx.x * blockDim.x + threadIdx.x; i < (long)Bv * Tv * Dv;
       i += stride) {
    unsigned short v = f2bf(x[i]);
    xb[i] = v;
    int d  = (int)(i % Dv);
    long bt = i / Dv;
    int t  = (int)(bt % Tv);
    int bb = (int)(bt / Tv);
    xT[((long)bb * Dv + d) * Tv + t] = v;
  }
}

__global__ void conv_wT_kernel(const float* __restrict__ W,
                               unsigned short* __restrict__ Wt, int K, int N) {
  long stride = (long)gridDim.x * blockDim.x;
  for (long i = (long)blockIdx.x * blockDim.x + threadIdx.x; i < (long)K * N;
       i += stride) {
    int k = (int)(i / N), n = (int)(i % N);
    Wt[(long)n * K + k] = f2bf(W[i]);
  }
}

// ======================= fused decoder kernel =======================
__global__ __launch_bounds__(NT, 1)
void decoder_fused(const unsigned short* __restrict__ xb,
                   const unsigned short* __restrict__ xT,
                   const unsigned short* __restrict__ w1t,
                   const unsigned short* __restrict__ w2t,
                   const float* __restrict__ b1, const float* __restrict__ b2,
                   float* __restrict__ out) {
  extern __shared__ char smem[];
  unsigned short* dec  = (unsigned short*)(smem + OFF_DEC);
  unsigned short* hls  = (unsigned short*)(smem + OFF_H);
  unsigned short* xkr0 = (unsigned short*)(smem + OFF_XKR0);
  unsigned short* xkc0 = (unsigned short*)(smem + OFF_XKC0);
  unsigned short* xkr1 = (unsigned short*)(smem + OFF_XKR1);
  unsigned short* xkc1 = (unsigned short*)(smem + OFF_XKC1);
  unsigned short* pbb  = (unsigned short*)(smem + OFF_P);
  unsigned short* wtA  = (unsigned short*)(smem + OFF_WA);
  unsigned short* wtB  = (unsigned short*)(smem + OFF_WB);
  const unsigned lds0 = (unsigned)(size_t)smem;  // LDS byte offset of dynamic base

  const int tid    = threadIdx.x;
  const int lane   = tid & 31;
  const int wv     = tid >> 5;
  const int b      = blockIdx.x >> 5;
  const int q0     = (blockIdx.x & 31) * ROWS;
  const int nlo    = lane & 15;
  const int rowOff = (lane >> 4) * 8;

  // ---- async staging helpers ----
  auto stage_keys = [&](int kb, int buf) {
    const unsigned lr = lds0 + (buf ? OFF_XKR1 : OFF_XKR0);
    const unsigned lc = lds0 + (buf ? OFF_XKC1 : OFF_XKC0);
    const unsigned short* gr = xb + ((long)b * Tv + kb * 32) * Dv;  // contiguous 16KB
    const unsigned short* gc = xT + (long)b * Dv * Tv + kb * 32;
#pragma unroll
    for (int i = 0; i < 8; ++i) {
      int f = i * NT + tid;                 // 0..1023 : 16B chunks
      async_b128(lr + f * 16, gr + f * 8);
      int d = f >> 2, srel = (f & 3) * 8;   // xT tile: rows d, 32 s contiguous
      async_b128(lc + (d * 32 + srel) * 2, gc + (long)d * Tv + srel);
    }
  };
  auto stage_w = [&](const unsigned short* Wt, int KW, int nc, int kc, int buf) {
    const unsigned lw = lds0 + (buf ? OFF_WB : OFF_WA);
#pragma unroll
    for (int i = 0; i < 8; ++i) {
      int f = i * NT + tid;
      int nl = f >> 2, krel = (f & 3) * 8;
      async_b128(lw + (nl * 32 + krel) * 2,
                 Wt + (long)(nc * 256 + nl) * KW + kc * 32 + krel);
    }
  };

  // ================= stage query rows: dec[:, 0:256] = x =================
  {
    const unsigned short* g = xb + ((long)b * Tv + q0) * Dv;  // contiguous 32KB
#pragma unroll
    for (int i = 0; i < 16; ++i) {
      int f = i * NT + tid;               // 0..2047 : 16B chunks
      int r = f >> 5, c = (f & 31) * 8;   // dec rows strided by 512
      async_b128(lds0 + OFF_DEC + (r * TWOD + c) * 2, g + f * 8);
    }
    wait_async0();
  }
  __syncthreads();

  // prefetch key chunk 0 while loading Q fragments
  stage_keys(0, 0);

  v16bf qf[8];
#pragma unroll
  for (int kc = 0; kc < 8; ++kc)
    qf[kc] = load_frag(dec + (wv * 16) * TWOD, TWOD, kc * 32, lane);

  v8f ctxa[16];
#pragma unroll
  for (int nt = 0; nt < 16; ++nt) ctxa[nt] = v8f_zero();
  float m[8], l[8];
#pragma unroll
  for (int v = 0; v < 8; ++v) { m[v] = -INFINITY; l[v] = 0.0f; }

  const float scale = 0.0625f;  // 1/sqrt(256)
  const int nkb = q0 / 32 + 2;
  unsigned short* pbw = pbb + wv * 16 * 32;

  // ================= flash attention (strictly causal) =================
  for (int kb = 0; kb < nkb; ++kb) {
    wait_async0();
    __syncthreads();                         // chunk kb resident in buf kb&1
    if (kb + 1 < nkb) stage_keys(kb + 1, (kb + 1) & 1);
    const unsigned short* xr = (kb & 1) ? xkr1 : xkr0;
    const unsigned short* xc = (kb & 1) ? xkc1 : xkc0;
    const int s0g = kb * 32;

    // S = Q * K^T : two 16x16 tiles over K-dim 256 (pipelined B-frag loads)
    v8f sA = v8f_zero(), sB = v8f_zero();
    v16bf k0 = load_frag(xr,           Dv, 0, lane);
    v16bf k1 = load_frag(xr + 16 * Dv, Dv, 0, lane);
#pragma unroll
    for (int kc = 0; kc < 8; ++kc) {
      v16bf c0 = k0, c1 = k1;
      if (kc < 7) {
        k0 = load_frag(xr,           Dv, (kc + 1) * 32, lane);
        k1 = load_frag(xr + 16 * Dv, Dv, (kc + 1) * 32, lane);
      }
      sA = wmma_bf16(qf[kc], c0, sA);
      sB = wmma_bf16(qf[kc], c1, sB);
    }

    // mask, online softmax update, emit P (bf16) to LDS
    float alpha[8];
#pragma unroll
    for (int v = 0; v < 8; ++v) {
      int tg = q0 + wv * 16 + v + rowOff;
      float a = sA[v] * scale, c = sB[v] * scale;
      if (s0g      + nlo >= tg) a = -1e9f;   // strictly causal: allow s < t
      if (s0g + 16 + nlo >= tg) c = -1e9f;
      float tm = fmaxf(a, c);
      tm = fmaxf(tm, __shfl_xor(tm, 1, 32));
      tm = fmaxf(tm, __shfl_xor(tm, 2, 32));
      tm = fmaxf(tm, __shfl_xor(tm, 4, 32));
      tm = fmaxf(tm, __shfl_xor(tm, 8, 32));
      float mn = fmaxf(m[v], tm);
      alpha[v] = __expf(m[v] - mn);
      m[v] = mn;
      float p0 = __expf(a - mn), p1 = __expf(c - mn);
      float rs = p0 + p1;
      rs += __shfl_xor(rs, 1, 32);
      rs += __shfl_xor(rs, 2, 32);
      rs += __shfl_xor(rs, 4, 32);
      rs += __shfl_xor(rs, 8, 32);
      l[v] = l[v] * alpha[v] + rs;
      pbw[(v + rowOff) * 32 + nlo]      = f2bf(p0);
      pbw[(v + rowOff) * 32 + nlo + 16] = f2bf(p1);
    }
#pragma unroll
    for (int nt = 0; nt < 16; ++nt)
#pragma unroll
      for (int v = 0; v < 8; ++v) ctxa[nt][v] *= alpha[v];
    __syncthreads();

    // ctx += P(16x32) * V(32x256), pipelined B-frags from col-major copy
    v16bf pf = load_frag(pbw, 32, 0, lane);
    v16bf vb = load_frag(xc, 32, 0, lane);
#pragma unroll
    for (int nt = 0; nt < 16; ++nt) {
      v16bf cur = vb;
      if (nt < 15) vb = load_frag(xc + (nt + 1) * 16 * 32, 32, 0, lane);
      ctxa[nt] = wmma_bf16(pf, cur, ctxa[nt]);
    }
    __syncthreads();   // done with this key buffer before it is restaged
  }

  // finalize: dec[:, 256:512] = ctx / l  (zeroed at t==0)
#pragma unroll
  for (int v = 0; v < 8; ++v) {
    int tg = q0 + wv * 16 + v + rowOff;
    float inv = (tg == 0) ? 0.0f : 1.0f / l[v];
#pragma unroll
    for (int nt = 0; nt < 16; ++nt) {
      float cv = ctxa[nt][v] * inv;
      dec[(wv * 16 + v + rowOff) * TWOD + Dv + nt * 16 + nlo] = f2bf(cv);
    }
  }
  __syncthreads();

  // ================= MLP layer 1: h = relu(dec @ W1 + b1) =================
  for (int nc = 0; nc < Hv / 256; ++nc) {
    v8f acc[16];
#pragma unroll
    for (int nt = 0; nt < 16; ++nt) acc[nt] = v8f_zero();
    stage_w(w1t, TWOD, nc, 0, 0);
    for (int kc = 0; kc < TWOD / 32; ++kc) {
      wait_async0();
      __syncthreads();
      if (kc + 1 < TWOD / 32) stage_w(w1t, TWOD, nc, kc + 1, (kc + 1) & 1);
      const unsigned short* wtp = (kc & 1) ? wtB : wtA;
      v16bf af = load_frag(dec + (wv * 16) * TWOD, TWOD, kc * 32, lane);
      v16bf bfr = load_frag(wtp, 32, 0, lane);
#pragma unroll
      for (int nt = 0; nt < 16; ++nt) {
        v16bf cur = bfr;
        if (nt < 15) bfr = load_frag(wtp + (nt + 1) * 16 * 32, 32, 0, lane);
        acc[nt] = wmma_bf16(af, cur, acc[nt]);
      }
      __syncthreads();
    }
#pragma unroll
    for (int nt = 0; nt < 16; ++nt) {
      int col = nc * 256 + nt * 16 + nlo;
      float bb = b1[col];
#pragma unroll
      for (int v = 0; v < 8; ++v) {
        float hv = fmaxf(acc[nt][v] + bb, 0.0f);
        hls[(wv * 16 + v + rowOff) * Hv + col] = f2bf(hv);
      }
    }
  }
  __syncthreads();

  // ================= MLP layer 2: out = h @ W2 + b2 (split halves) =================
  for (int nc = 0; nc < OUT2v / 256; ++nc) {
    v8f acc[16];
#pragma unroll
    for (int nt = 0; nt < 16; ++nt) acc[nt] = v8f_zero();
    stage_w(w2t, Hv, nc, 0, 0);
    for (int kc = 0; kc < Hv / 32; ++kc) {
      wait_async0();
      __syncthreads();
      if (kc + 1 < Hv / 32) stage_w(w2t, Hv, nc, kc + 1, (kc + 1) & 1);
      const unsigned short* wtp = (kc & 1) ? wtB : wtA;
      v16bf af = load_frag(hls + (wv * 16) * Hv, Hv, kc * 32, lane);
      v16bf bfr = load_frag(wtp, 32, 0, lane);
#pragma unroll
      for (int nt = 0; nt < 16; ++nt) {
        v16bf cur = bfr;
        if (nt < 15) bfr = load_frag(wtp + (nt + 1) * 16 * 32, 32, 0, lane);
        acc[nt] = wmma_bf16(af, cur, acc[nt]);
      }
      __syncthreads();
    }
    // out tuple is concatenated flat: [B*T*256 | B*T*256]; nc selects the half
#pragma unroll
    for (int nt = 0; nt < 16; ++nt) {
      int cc = nt * 16 + nlo;
      float bb = b2[nc * 256 + cc];
#pragma unroll
      for (int v = 0; v < 8; ++v) {
        int rowg = b * Tv + q0 + wv * 16 + v + rowOff;
        out[(long)nc * (Bv * Tv * 256) + (long)rowg * 256 + cc] = acc[nt][v] + bb;
      }
    }
  }
}

extern "C" void kernel_launch(void* const* d_in, const int* in_sizes, int n_in,
                              void* d_out, int out_size, void* d_ws, size_t ws_size,
                              hipStream_t stream) {
  const float* x  = (const float*)d_in[0];
  const float* W1 = (const float*)d_in[1];
  const float* b1 = (const float*)d_in[2];
  const float* W2 = (const float*)d_in[3];
  const float* b2 = (const float*)d_in[4];
  float* out = (float*)d_out;

  unsigned short* ws = (unsigned short*)d_ws;
  unsigned short* xb  = ws + WS_XB;
  unsigned short* xT  = ws + WS_XT;
  unsigned short* w1t = ws + WS_W1T;
  unsigned short* w2t = ws + WS_W2T;

  (void)in_sizes; (void)n_in; (void)out_size; (void)ws_size;

  // prep: one-time f32->bf16 conversion (+ transposed copies for B-operand tiles)
  conv_x_kernel<<<4096, 256, 0, stream>>>(x, xb, xT);
  conv_wT_kernel<<<1024, 256, 0, stream>>>(W1, w1t, TWOD, Hv);
  conv_wT_kernel<<<1024, 256, 0, stream>>>(W2, w2t, Hv, OUT2v);

  hipFuncSetAttribute((const void*)decoder_fused,
                      hipFuncAttributeMaxDynamicSharedMemorySize, SMEM_BYTES);
  dim3 grid(Bv * (Tv / ROWS));  // 1024 blocks
  dim3 block(NT);               // 4 wave32s
  decoder_fused<<<grid, block, SMEM_BYTES, stream>>>(xb, xT, w1t, w2t, b1, b2, out);
}
